// MiniCPMAttention_34643206210202
// MI455X (gfx1250) — matrix operand order repaired and checked by reading, not compiled, not run on hardware
//
#include <hip/hip_runtime.h>
#include <hip/hip_bf16.h>

// ---------------------------------------------------------------------------
// MiniCPM attention layer for gfx1250 (MI455X)
//   - wave32 WMMA f16 (v_wmma_f32_16x16x32_f16) for QKV GEMM, attention, O-proj
//   - double-buffered LDS GEMM fed by GLOBAL_LOAD_ASYNC_TO_LDS_B128 (ASYNCcnt)
//   - flash-attention with global_prefetch of the next K/V block
// ---------------------------------------------------------------------------

typedef _Float16 f16;
typedef __attribute__((ext_vector_type(8)))  _Float16 f16x8;
typedef __attribute__((ext_vector_type(16))) _Float16 f16x16;
typedef __attribute__((ext_vector_type(8)))  float    f32x8;

union Frag {            // one WMMA A/B operand: 16 f16 per lane = 8 VGPRs
  f16x16 v;
  f16x8  h[2];
};

#define HIDDEN 2048
#define NH     32
#define NKV    8
#define HD     64
#define SEQ    2048
#define BATCH  2
#define QKV_N  3072            // (32 + 2*8) * 64
#define BS     (BATCH*SEQ)     // 4096
#define TPAD   40              // 32-deep K tile padded to 40 f16 (bank-conflict free)
#define PPAD   72              // 64-wide P tile padded to 72 f16
#define BM     128
#define BN     256
#define BK     32

static __device__ __forceinline__ f32x8 wmma16(const f16x16& a, const f16x16& b, f32x8 c) {
  return __builtin_amdgcn_wmma_f32_16x16x32_f16(false, a, false, b, (short)0, c, false, false);
}

// Async DMA: global -> LDS, 16 B per lane, tracked by ASYNCcnt (CDNA5 §15.18.3)
static __device__ __forceinline__ void async_ld_b128(unsigned lds_off, const f16* g) {
  asm volatile("global_load_async_to_lds_b128 %0, %1, off"
               :: "v"(lds_off), "v"((unsigned long long)g)
               : "memory");
}
static __device__ __forceinline__ void wait_async_le6() {
  asm volatile("s_wait_asynccnt 0x6" ::: "memory");
}
static __device__ __forceinline__ void wait_async_0() {
  asm volatile("s_wait_asynccnt 0x0" ::: "memory");
}

// -------------------------- precision / layout prep -------------------------

__global__ void cvt_f32_to_f16_kernel(const float* __restrict__ in,
                                      f16* __restrict__ out, int n) {
  int i = blockIdx.x * blockDim.x + threadIdx.x;
  if (i < n) out[i] = (f16)in[i];
}

// in: rows x cols (row-major f32)  ->  out: cols x rows (row-major f16)
__global__ void transpose_to_f16_kernel(const float* __restrict__ in,
                                        f16* __restrict__ out, int rows, int cols) {
  int i = blockIdx.x * blockDim.x + threadIdx.x;
  if (i >= rows * cols) return;
  int r = i / cols, c = i - r * cols;
  out[(size_t)c * rows + r] = (f16)in[i];
}

// ------------------------------- tiled GEMM --------------------------------
// C[M,N] = A[M,K] x B[K,N];  A row-major f16, Bt = B^T (N x K) row-major f16.
// Block tile 128x256, 8 waves (2x4), wave tile 64x64 = 4x4 WMMA frags, K-step 32.
// Double-buffered LDS, staged with async global->LDS DMA (6 ops/wave/tile).
__global__ __launch_bounds__(256)
void gemm_f16_tn_kernel(const f16* __restrict__ A, const f16* __restrict__ Bt,
                        float* __restrict__ C, int M, int N, int K) {
  __shared__ __align__(16) f16 As[2][BM * TPAD];
  __shared__ __align__(16) f16 Bs[2][BN * TPAD];

  const int tid  = threadIdx.x;
  const int lane = tid & 31;
  const int l15  = lane & 15;
  const int hi16 = lane >> 4;          // 0 or 1 (half-wave select)
  const int w    = tid >> 5;           // wave 0..7
  const int wm   = w >> 2;             // 0..1  -> 64-row slice
  const int wn   = w & 3;              // 0..3  -> 64-col slice
  const int m0   = blockIdx.y * BM;
  const int n0   = blockIdx.x * BN;

  // per-thread staging: A = 512 16B-chunks (2/thread), B = 1024 (4/thread)
  auto stage = [&](int kt, int p) {
    unsigned aBase = (unsigned)(size_t)&As[p][0];
    unsigned bBase = (unsigned)(size_t)&Bs[p][0];
    #pragma unroll
    for (int j = 0; j < 2; ++j) {
      int c = tid + j * 256;
      int row = c >> 2, ko = (c & 3) << 3;
      async_ld_b128(aBase + (unsigned)(row * TPAD + ko) * 2,
                    &A[(size_t)(m0 + row) * K + kt + ko]);
    }
    #pragma unroll
    for (int j = 0; j < 4; ++j) {
      int c = tid + j * 256;
      int row = c >> 2, ko = (c & 3) << 3;
      async_ld_b128(bBase + (unsigned)(row * TPAD + ko) * 2,
                    &Bt[(size_t)(n0 + row) * K + kt + ko]);
    }
  };

  f32x8 acc[4][4] = {};

  stage(0, 0);
  const int kTiles = K / BK;
  for (int i = 0; i < kTiles; ++i) {
    const int p = i & 1;
    if (i + 1 < kTiles) {
      stage((i + 1) * BK, p ^ 1);   // overlap next tile's DMA with this compute
      wait_async_le6();             // in-order completion => tile i resident
    } else {
      wait_async_0();
    }
    __syncthreads();

    Frag a[4];
    #pragma unroll
    for (int mf = 0; mf < 4; ++mf) {          // A frag: lane holds M=l15, K 0..7 & 16..23 (+8 hi half)
      int base = (wm * 64 + mf * 16 + l15) * TPAD + (hi16 << 3);
      a[mf].h[0] = *(const f16x8*)&As[p][base];
      a[mf].h[1] = *(const f16x8*)&As[p][base + 16];
    }
    #pragma unroll
    for (int nf = 0; nf < 4; ++nf) {          // B frag: lane holds N=l15, K 0..7 & 16..23 (+8 hi half)
      Frag b;
      int base = (wn * 64 + nf * 16 + l15) * TPAD + (hi16 << 3);
      b.h[0] = *(const f16x8*)&Bs[p][base];
      b.h[1] = *(const f16x8*)&Bs[p][base + 16];
      #pragma unroll
      for (int mf = 0; mf < 4; ++mf)
        acc[mf][nf] = wmma16(a[mf].v, b.v, acc[mf][nf]);
    }
    __syncthreads();                // reads done before next iter's DMA overwrites
  }

  // D layout: lane = N (mod 16); VGPR r = row r (lanes 0-15) / row r+8 (lanes 16-31)
  #pragma unroll
  for (int mf = 0; mf < 4; ++mf)
    #pragma unroll
    for (int nf = 0; nf < 4; ++nf)
      #pragma unroll
      for (int r = 0; r < 8; ++r) {
        int row = m0 + wm * 64 + mf * 16 + r + (hi16 << 3);
        int col = n0 + wn * 64 + nf * 16 + l15;
        C[(size_t)row * N + col] = acc[mf][nf][r];
      }
}

// ------------------------------ RoPE + split -------------------------------
// qkv (BS x 3072 f32) -> Q (B,NH,S,HD f16, pre-scaled by d^-1/2), K (B,NKV,S,HD f16)
__global__ void rope_split_kernel(const int* __restrict__ pos,
                                  const float* __restrict__ qkv,
                                  f16* __restrict__ qb, f16* __restrict__ kb) {
  int i = blockIdx.x * blockDim.x + threadIdx.x;
  if (i >= BS * 40 * 32) return;
  int d  = i & 31;
  int t  = i >> 5;
  int hs = t % 40;  t /= 40;
  int s  = t % SEQ;
  int b  = t / SEQ;

  float p    = (float)pos[b * SEQ + s];
  float freq = p * __expf((float)d * (-9.210340371976184f / 32.0f)); // theta^(-d/32)
  float sn, cs;
  __sincosf(freq, &sn, &cs);

  const float* row = qkv + (size_t)(b * SEQ + s) * QKV_N;
  if (hs < NH) {
    float x1 = row[hs * HD + d], x2 = row[hs * HD + d + 32];
    f16* q = qb + ((size_t)(b * NH + hs) * SEQ + s) * HD;
    q[d]      = (f16)((x1 * cs - x2 * sn) * 0.125f);   // fold in 1/sqrt(64)
    q[d + 32] = (f16)((x2 * cs + x1 * sn) * 0.125f);
  } else {
    int h = hs - NH;
    float x1 = row[2048 + h * HD + d], x2 = row[2048 + h * HD + d + 32];
    f16* k = kb + ((size_t)(b * NKV + h) * SEQ + s) * HD;
    k[d]      = (f16)(x1 * cs - x2 * sn);
    k[d + 32] = (f16)(x2 * cs + x1 * sn);
  }
}

// V -> V^T (B,NKV,HD,SEQ) f16, so P@V B-fragments are contiguous along seq
__global__ void v_transpose_kernel(const float* __restrict__ qkv,
                                   f16* __restrict__ vt) {
  int i = blockIdx.x * blockDim.x + threadIdx.x;
  if (i >= BS * NKV * HD) return;
  int d = i & 63;
  int t = i >> 6;
  int h = t % NKV; t /= NKV;
  int s = t % SEQ;
  int b = t / SEQ;
  vt[((size_t)(b * NKV + h) * HD + d) * SEQ + s] =
      (f16)qkv[(size_t)(b * SEQ + s) * QKV_N + 2560 + h * HD + d];
}

// ---------------------------- flash attention ------------------------------
// grid = (SEQ/128, NH, BATCH); 8 waves/block, each wave owns 16 Q rows.
// Streams causal K/V in 64-wide blocks. Q@K^T and P@V both on WMMA f16.
__global__ __launch_bounds__(256)
void flash_attn_kernel(const f16* __restrict__ qb, const f16* __restrict__ kbuf,
                       const f16* __restrict__ vt, f16* __restrict__ attn) {
  __shared__ __align__(16) f16 pbuf[8][16 * PPAD];

  const int lane = threadIdx.x & 31;
  const int l15  = lane & 15;
  const int hi16 = lane >> 4;
  const int w    = threadIdx.x >> 5;
  const int h    = blockIdx.y;
  const int b    = blockIdx.z;
  const int hk   = h >> 2;                     // GQA: 4 Q heads per KV head
  const int q0   = blockIdx.x * 128 + w * 16;

  const f16* kh = kbuf + (size_t)(b * NKV + hk) * SEQ * HD;   // K rows, 128 B each
  const f16* vh = vt   + (size_t)(b * NKV + hk) * HD * SEQ;   // V^T rows, 4 KB each

  // Q fragments (persist across the whole K sweep): A-matrix layout
  const f16* qrow = qb + ((size_t)(b * NH + h) * SEQ + q0 + l15) * HD;
  Frag qa[2];
  #pragma unroll
  for (int t = 0; t < 2; ++t) {
    qa[t].h[0] = *(const f16x8*)(qrow + t * 32 + (hi16 << 3));
    qa[t].h[1] = *(const f16x8*)(qrow + t * 32 + (hi16 << 3) + 16);
  }

  f32x8 o[4] = {};
  float mrun[8], lrun[8];
  #pragma unroll
  for (int r = 0; r < 8; ++r) { mrun[r] = -3.0e38f; lrun[r] = 0.0f; }

  const int nblk = (q0 + 15) / 64 + 1;         // causal: only blocks touching rows
  for (int blk = 0; blk < nblk; ++blk) {
    const int kb = blk * 64;

    // prefetch next K block (64 rows x 128 B) and V^T columns while we compute
    if (blk + 1 < nblk) {
      const f16* kn = kh + (size_t)(kb + 64 + lane * 2) * HD;
      __builtin_prefetch(kn, 0, 3);
      __builtin_prefetch(kn + HD, 0, 3);
      const f16* vn = vh + (size_t)(lane * 2) * SEQ + kb + 64;
      __builtin_prefetch(vn, 0, 3);
      __builtin_prefetch(vn + SEQ, 0, 3);
    }

    // ---- scores: S = Q @ K^T (K rows are exactly the B^T rows we need) ----
    f32x8 sc[4];
    #pragma unroll
    for (int nf = 0; nf < 4; ++nf) {
      const f16* krow = kh + (size_t)(kb + nf * 16 + l15) * HD;
      Frag kf[2];
      #pragma unroll
      for (int t = 0; t < 2; ++t) {
        kf[t].h[0] = *(const f16x8*)(krow + t * 32 + (hi16 << 3));
        kf[t].h[1] = *(const f16x8*)(krow + t * 32 + (hi16 << 3) + 16);
      }
      f32x8 s = {};
      s = wmma16(qa[0].v, kf[0].v, s);
      s = wmma16(qa[1].v, kf[1].v, s);
      sc[nf] = s;
    }

    // ---- causal mask + online softmax (row = r + 8*hi16, col = lane) ----
    float mblk[8];
    #pragma unroll
    for (int r = 0; r < 8; ++r) mblk[r] = -3.0e38f;
    #pragma unroll
    for (int nf = 0; nf < 4; ++nf)
      #pragma unroll
      for (int r = 0; r < 8; ++r) {
        int col = kb + nf * 16 + l15;
        int row = q0 + r + (hi16 << 3);
        float v = sc[nf][r];
        if (col > row) v = -3.0e38f;
        sc[nf][r] = v;
        mblk[r] = fmaxf(mblk[r], v);
      }
    #pragma unroll
    for (int m = 8; m >= 1; m >>= 1)           // reduce within 16-lane row group
      #pragma unroll
      for (int r = 0; r < 8; ++r)
        mblk[r] = fmaxf(mblk[r], __shfl_xor(mblk[r], m, 32));

    float corr[8];
    #pragma unroll
    for (int r = 0; r < 8; ++r) {
      float mn = fmaxf(mrun[r], mblk[r]);
      corr[r]  = __expf(mrun[r] - mn);
      mrun[r]  = mn;
      lrun[r] *= corr[r];
    }
    #pragma unroll
    for (int of = 0; of < 4; ++of)
      #pragma unroll
      for (int r = 0; r < 8; ++r) o[of][r] *= corr[r];

    float lblk[8];
    #pragma unroll
    for (int r = 0; r < 8; ++r) lblk[r] = 0.0f;
    #pragma unroll
    for (int nf = 0; nf < 4; ++nf)
      #pragma unroll
      for (int r = 0; r < 8; ++r) {
        float p = __expf(sc[nf][r] - mrun[r]);
        lblk[r] += p;
        pbuf[w][(r + (hi16 << 3)) * PPAD + nf * 16 + l15] = (f16)p;  // transpose via LDS
      }
    #pragma unroll
    for (int m = 8; m >= 1; m >>= 1)
      #pragma unroll
      for (int r = 0; r < 8; ++r)
        lblk[r] += __shfl_xor(lblk[r], m, 32);
    #pragma unroll
    for (int r = 0; r < 8; ++r) lrun[r] += lblk[r];

    // ---- O += P @ V  (A = P from LDS, B^T rows = V^T rows, contiguous) ----
    Frag pa[2];
    #pragma unroll
    for (int t = 0; t < 2; ++t) {
      int base = l15 * PPAD + t * 32 + (hi16 << 3);
      pa[t].h[0] = *(const f16x8*)&pbuf[w][base];
      pa[t].h[1] = *(const f16x8*)&pbuf[w][base + 16];
    }
    #pragma unroll
    for (int of = 0; of < 4; ++of) {
      const f16* vrow = vh + (size_t)(of * 16 + l15) * SEQ + kb;
      Frag vf[2];
      #pragma unroll
      for (int t = 0; t < 2; ++t) {
        vf[t].h[0] = *(const f16x8*)(vrow + t * 32 + (hi16 << 3));
        vf[t].h[1] = *(const f16x8*)(vrow + t * 32 + (hi16 << 3) + 16);
      }
      o[of] = wmma16(pa[0].v, vf[0].v, o[of]);
      o[of] = wmma16(pa[1].v, vf[1].v, o[of]);
    }
  }

  // ---- finalize: O / l, write (B,S,NH*HD) f16 for the output projection ----
  #pragma unroll
  for (int r = 0; r < 8; ++r) lrun[r] = 1.0f / lrun[r];
  #pragma unroll
  for (int of = 0; of < 4; ++of)
    #pragma unroll
    for (int r = 0; r < 8; ++r) {
      int row = q0 + r + (hi16 << 3);
      attn[((size_t)b * SEQ + row) * HIDDEN + h * HD + of * 16 + l15] =
          (f16)(o[of][r] * lrun[r]);
    }
}

// ------------------------------- launcher ----------------------------------

static inline size_t alignup256(size_t x) { return (x + 255) & ~(size_t)255; }

extern "C" void kernel_launch(void* const* d_in, const int* in_sizes, int n_in,
                              void* d_out, int out_size, void* d_ws, size_t ws_size,
                              hipStream_t stream) {
  (void)in_sizes; (void)n_in; (void)out_size; (void)ws_size;

  const int*   positions = (const int*)d_in[0];
  const float* hidden    = (const float*)d_in[1];
  const float* w_qkv     = (const float*)d_in[2];
  const float* w_o       = (const float*)d_in[3];
  float*       out       = (float*)d_out;

  char* p = (char*)d_ws;
  f16*   hid16 = (f16*)p;   p += alignup256((size_t)BS * HIDDEN * 2);
  f16*   wqkvt = (f16*)p;   p += alignup256((size_t)QKV_N * HIDDEN * 2);
  f16*   wot   = (f16*)p;   p += alignup256((size_t)HIDDEN * HIDDEN * 2);
  float* qkvf  = (float*)p; p += alignup256((size_t)BS * QKV_N * 4);
  f16*   qb    = (f16*)p;   p += alignup256((size_t)BATCH * NH  * SEQ * HD * 2);
  f16*   kb    = (f16*)p;   p += alignup256((size_t)BATCH * NKV * SEQ * HD * 2);
  f16*   vtb   = (f16*)p;   p += alignup256((size_t)BATCH * NKV * SEQ * HD * 2);
  f16*   attn  = (f16*)p;   p += alignup256((size_t)BS * HIDDEN * 2);

  // 1) precision/layout prep
  {
    int n = BS * HIDDEN;
    cvt_f32_to_f16_kernel<<<(n + 255) / 256, 256, 0, stream>>>(hidden, hid16, n);
  }
  {
    int n = HIDDEN * QKV_N;
    transpose_to_f16_kernel<<<(n + 255) / 256, 256, 0, stream>>>(w_qkv, wqkvt, HIDDEN, QKV_N);
  }
  {
    int n = HIDDEN * HIDDEN;
    transpose_to_f16_kernel<<<(n + 255) / 256, 256, 0, stream>>>(w_o, wot, HIDDEN, HIDDEN);
  }

  // 2) QKV projection (4096x2048 @ 2048x3072)
  gemm_f16_tn_kernel<<<dim3(QKV_N / BN, BS / BM), 256, 0, stream>>>(
      hid16, wqkvt, qkvf, BS, QKV_N, HIDDEN);

  // 3) RoPE + head split (+scale), V transpose
  {
    int n = BS * 40 * 32;
    rope_split_kernel<<<(n + 255) / 256, 256, 0, stream>>>(positions, qkvf, qb, kb);
  }
  {
    int n = BS * NKV * HD;
    v_transpose_kernel<<<(n + 255) / 256, 256, 0, stream>>>(qkvf, vtb);
  }

  // 4) causal GQA flash attention
  flash_attn_kernel<<<dim3(SEQ / 128, NH, BATCH), 256, 0, stream>>>(qb, kb, vtb, attn);

  // 5) output projection (4096x2048 @ 2048x2048) -> f32 d_out
  gemm_f16_tn_kernel<<<dim3(HIDDEN / BN, BS / BM), 256, 0, stream>>>(
      attn, wot, out, BS, HIDDEN, HIDDEN);
}
